// SmallGAT_77584289235225
// MI455X (gfx1250) — matrix operand order repaired and checked by reading, not compiled, not run on hardware
//
#include <hip/hip_runtime.h>
#include <hip/hip_bf16.h>

typedef __attribute__((ext_vector_type(16))) _Float16 v16h;
typedef __attribute__((ext_vector_type(8)))  float    v8f;

#define HIDDEN 128
#define HEADS 2
#define CPH 64
#define NCLASSES 32
#define NGRAPHS 64
#define NEG_SLOPE 0.2f

// ---------- helpers: order-preserving float <-> uint for atomicMax ----------
__device__ __forceinline__ unsigned encf(float f) {
    unsigned u = __float_as_uint(f);
    return (u & 0x80000000u) ? ~u : (u | 0x80000000u);
}
__device__ __forceinline__ float decf(unsigned u) {
    return (u & 0x80000000u) ? __uint_as_float(u ^ 0x80000000u) : __uint_as_float(~u);
}

// ---------- fp32 -> fp16 weight conversion ----------
__global__ void cvt_f32_to_f16(const float* __restrict__ in, _Float16* __restrict__ out, int n) {
    int i = blockIdx.x * blockDim.x + threadIdx.x;
    if (i < n) out[i] = (_Float16)in[i];
}

// ---------- WMMA GEMM: out[N,128] = A[N,128] @ W[128,128]^T ----------
// One wave computes a 16-row strip; 8 waves per block -> 128 rows per block.
// Fragment layouts used by this kernel (wave32, contiguous k order in A and B):
//   A 16x32 f16 : lane l -> row m=l&15; halves j   = K[kh*16+j]
//   B 32x16 f16 : lane l -> col n=l&15; halves j   = K[kh*16+j]  (contiguous W row slice)
//   D 16x16 f32 : elem v -> row rowBase + v + 8*kh, col = l&15
__global__ __launch_bounds__(256) void gemm_xw_wmma(
    const float* __restrict__ A, const _Float16* __restrict__ W16,
    float* __restrict__ out, int N)
{
    const int wave = threadIdx.x >> 5;
    const int lane = threadIdx.x & 31;
    const int rowBase = blockIdx.x * 128 + wave * 16;
    const int m  = lane & 15;
    const int kh = lane >> 4;

    const bool fullTile = (rowBase + 16) <= N;   // wave-uniform
    int row  = rowBase + m;
    int rowc = (row < N) ? row : 0;              // clamp (garbage rows never stored)

    // A fragments for K = 0..127 (4 k-steps of 32), f32 -> f16 inline.
    v16h afr[4];
#pragma unroll
    for (int ks = 0; ks < 4; ++ks) {
        const float* p0 = A + (size_t)rowc * HIDDEN + ks * 32 + kh * 16;
        const float* p1 = p0 + 8;
#pragma unroll
        for (int j = 0; j < 8; ++j) {
            afr[ks][j]     = (_Float16)p0[j];
            afr[ks][8 + j] = (_Float16)p1[j];
        }
    }

    const int n = lane & 15;
#pragma unroll
    for (int nt = 0; nt < 8; ++nt) {
        const int o = nt * 16 + n;               // output channel = W row (contiguous K)
        v8f acc = {};
#pragma unroll
        for (int ks = 0; ks < 4; ++ks) {
            v16h bfr;
            const _Float16* q = W16 + (size_t)o * HIDDEN + ks * 32 + kh * 16;
#pragma unroll
            for (int j = 0; j < 16; ++j) bfr[j] = q[j];
            acc = __builtin_amdgcn_wmma_f32_16x16x32_f16(
                false, afr[ks], false, bfr, (short)0, acc, false, false);
        }
        float* obase = out + (size_t)(rowBase + 8 * kh) * HIDDEN + nt * 16 + n;
        if (fullTile) {
            // uniform path: straight-line coalesced stores, no exec juggling
#pragma unroll
            for (int v = 0; v < 8; ++v) obase[(size_t)v * HIDDEN] = acc[v];
        } else {
#pragma unroll
            for (int v = 0; v < 8; ++v) {
                if (rowBase + 8 * kh + v < N) obase[(size_t)v * HIDDEN] = acc[v];
            }
        }
    }
}

// ---------- per-(node,head) attention dots ----------
__global__ void attn_dots(const float* __restrict__ xl,
                          const float* __restrict__ att_src, const float* __restrict__ att_dst,
                          float* __restrict__ asrc, float* __restrict__ adst, int N)
{
    int i = blockIdx.x * blockDim.x + threadIdx.x;
    if (i >= N * HEADS) return;
    int nnode = i >> 1, h = i & 1;
    const float* xr = xl + (size_t)nnode * HIDDEN + h * CPH;
    const float* as = att_src + h * CPH;
    const float* ad = att_dst + h * CPH;
    float s = 0.f, d = 0.f;
#pragma unroll 4
    for (int c = 0; c < CPH; ++c) { float v = xr[c]; s += v * as[c]; d += v * ad[c]; }
    asrc[i] = s; adst[i] = d;
}

// ---------- phase 1: segment max over dst ----------
__global__ void edge_max(const int* __restrict__ ei, int E, int N,
                         const float* __restrict__ asrc, const float* __restrict__ adst,
                         unsigned* __restrict__ amax)
{
    int e = blockIdx.x * blockDim.x + threadIdx.x;
    if (e >= E + N) return;
    int s, d;
    if (e < E) { s = ei[e]; d = ei[E + e]; } else { s = d = e - E; }
#pragma unroll
    for (int h = 0; h < HEADS; ++h) {
        float a = asrc[s * 2 + h] + adst[d * 2 + h];
        a = (a > 0.f) ? a : NEG_SLOPE * a;
        atomicMax(&amax[d * 2 + h], encf(a));
    }
}

// ---------- phase 2: exp & denominator ----------
__global__ void edge_exp(const int* __restrict__ ei, int E, int N,
                         const float* __restrict__ asrc, const float* __restrict__ adst,
                         const unsigned* __restrict__ amax,
                         float* __restrict__ denom, float* __restrict__ ex)
{
    int e = blockIdx.x * blockDim.x + threadIdx.x;
    if (e >= E + N) return;
    int s, d;
    if (e < E) { s = ei[e]; d = ei[E + e]; } else { s = d = e - E; }
#pragma unroll
    for (int h = 0; h < HEADS; ++h) {
        float a = asrc[s * 2 + h] + adst[d * 2 + h];
        a = (a > 0.f) ? a : NEG_SLOPE * a;
        float v = __expf(a - decf(amax[d * 2 + h]));
        ex[(size_t)e * 2 + h] = v;
        atomicAdd(&denom[d * 2 + h], v);
    }
}

// ---------- phase 3: weighted scatter (edge x 4-channel chunk) ----------
__global__ void edge_scatter(const int* __restrict__ ei, int E, int N,
                             const float* __restrict__ ex, const float* __restrict__ denom,
                             const float* __restrict__ xl, float* __restrict__ out)
{
    long long t = (long long)blockIdx.x * blockDim.x + threadIdx.x;
    if (t >= (long long)(E + N) * 32) return;
    int e = (int)(t >> 5);
    int chunk = (int)(t & 31);
    int s, d;
    if (e < E) { s = ei[e]; d = ei[E + e]; } else { s = d = e - E; }
    int c0 = chunk * 4;
    int h = c0 >> 6;
    float dn = denom[d * 2 + h];
    dn = (dn > 1e-16f) ? dn : 1e-16f;
    float coeff = ex[(size_t)e * 2 + h] / dn;
    const float4 v = *(const float4*)(xl + (size_t)s * HIDDEN + c0);
    float* o = out + (size_t)d * HIDDEN + c0;
    atomicAdd(o + 0, v.x * coeff);
    atomicAdd(o + 1, v.y * coeff);
    atomicAdd(o + 2, v.z * coeff);
    atomicAdd(o + 3, v.w * coeff);
}

// ---------- bias + relu (in place) ----------
__global__ void bias_relu(float* __restrict__ buf, const float* __restrict__ b, int N)
{
    long long i = (long long)blockIdx.x * blockDim.x + threadIdx.x;
    if (i >= (long long)N * HIDDEN) return;
    int c = (int)(i & (HIDDEN - 1));
    float v = buf[i] + b[c];
    buf[i] = (v > 0.f) ? v : 0.f;
}

// ---------- global mean pool (sum + counts) ----------
__global__ void pool_sum(const float* __restrict__ h, const int* __restrict__ batch,
                         float* __restrict__ pooled, unsigned* __restrict__ counts, int N)
{
    long long i = (long long)blockIdx.x * blockDim.x + threadIdx.x;
    if (i >= (long long)N * HIDDEN) return;
    int nn = (int)(i >> 7);
    int c  = (int)(i & (HIDDEN - 1));
    int g = batch[nn];
    atomicAdd(&pooled[(size_t)g * HIDDEN + c], h[i]);
    if (c == 0) atomicAdd(&counts[g], 1u);
}

// ---------- final FC: [64,128] -> [64,32] ----------
__global__ void fc_out(const float* __restrict__ pooled, const unsigned* __restrict__ counts,
                       const float* __restrict__ fw, const float* __restrict__ fb,
                       float* __restrict__ out)
{
    int i = blockIdx.x * blockDim.x + threadIdx.x;
    if (i >= NGRAPHS * NCLASSES) return;
    int g = i >> 5, o = i & 31;
    float cnt = counts[g] ? (float)counts[g] : 1.f;
    const float* pr = pooled + (size_t)g * HIDDEN;
    const float* wr = fw + (size_t)o * HIDDEN;
    float s = 0.f;
#pragma unroll 4
    for (int c = 0; c < HIDDEN; ++c) s += pr[c] * wr[c];
    out[i] = s / cnt + fb[o];   // (sum/cnt) dot w == sum dot w / cnt since cnt uniform per row
}

extern "C" void kernel_launch(void* const* d_in, const int* in_sizes, int n_in,
                              void* d_out, int out_size, void* d_ws, size_t ws_size,
                              hipStream_t stream) {
    const float* x        = (const float*)d_in[0];
    const int*   ei       = (const int*)d_in[1];
    const int*   batch    = (const int*)d_in[2];
    const float* W1       = (const float*)d_in[3];
    const float* att_src1 = (const float*)d_in[4];
    const float* att_dst1 = (const float*)d_in[5];
    const float* b1       = (const float*)d_in[6];
    const float* W2       = (const float*)d_in[7];
    const float* att_src2 = (const float*)d_in[8];
    const float* att_dst2 = (const float*)d_in[9];
    const float* b2       = (const float*)d_in[10];
    const float* fc_w     = (const float*)d_in[11];
    const float* fc_b     = (const float*)d_in[12];
    float* out = (float*)d_out;

    const int N = in_sizes[2];          // 50000 nodes
    const int E = in_sizes[1] / 2;      // 800000 edges
    const int ET = E + N;               // with self loops

    // workspace layout (256B aligned)
    char* ws = (char*)d_ws;
    size_t off = 0;
    auto take = [&](size_t bytes) -> char* {
        char* p = ws + off;
        off = (off + bytes + 255) & ~(size_t)255;
        return p;
    };
    float*    bufA   = (float*)take((size_t)N * HIDDEN * 4);      // xl (layer input to messages)
    float*    bufB   = (float*)take((size_t)N * HIDDEN * 4);      // message accumulator / h
    _Float16* w16_1  = (_Float16*)take(HIDDEN * HIDDEN * 2);
    _Float16* w16_2  = (_Float16*)take(HIDDEN * HIDDEN * 2);
    float*    asrc   = (float*)take((size_t)N * HEADS * 4);
    float*    adst   = (float*)take((size_t)N * HEADS * 4);
    unsigned* amax   = (unsigned*)take((size_t)N * HEADS * 4);
    float*    denom  = (float*)take((size_t)N * HEADS * 4);
    float*    exbuf  = (float*)take((size_t)ET * HEADS * 4);
    float*    pooled = (float*)take((size_t)NGRAPHS * HIDDEN * 4);
    unsigned* counts = (unsigned*)take((size_t)NGRAPHS * 4);
    (void)ws_size; (void)n_in;

    // convert weights to f16 once per call
    cvt_f32_to_f16<<<(HIDDEN * HIDDEN + 255) / 256, 256, 0, stream>>>(W1, w16_1, HIDDEN * HIDDEN);
    cvt_f32_to_f16<<<(HIDDEN * HIDDEN + 255) / 256, 256, 0, stream>>>(W2, w16_2, HIDDEN * HIDDEN);

    const int gemmBlocks   = (N + 127) / 128;
    const int nodeHBlocks  = (N * HEADS + 255) / 256;
    const int edgeBlocks   = (ET + 255) / 256;
    const long long scatT  = (long long)ET * 32;
    const int scatBlocks   = (int)((scatT + 255) / 256);
    const long long ncT    = (long long)N * HIDDEN;
    const int ncBlocks     = (int)((ncT + 255) / 256);

    for (int layer = 0; layer < 2; ++layer) {
        const float*    gin  = (layer == 0) ? x : bufB;
        const _Float16* w16  = (layer == 0) ? w16_1 : w16_2;
        const float*    asv  = (layer == 0) ? att_src1 : att_src2;
        const float*    adv  = (layer == 0) ? att_dst1 : att_dst2;
        const float*    bias = (layer == 0) ? b1 : b2;

        gemm_xw_wmma<<<gemmBlocks, 256, 0, stream>>>(gin, w16, bufA, N);
        attn_dots<<<nodeHBlocks, 256, 0, stream>>>(bufA, asv, adv, asrc, adst, N);

        hipMemsetAsync(amax,  0, (size_t)N * HEADS * 4, stream);
        hipMemsetAsync(denom, 0, (size_t)N * HEADS * 4, stream);

        edge_max<<<edgeBlocks, 256, 0, stream>>>(ei, E, N, asrc, adst, amax);
        edge_exp<<<edgeBlocks, 256, 0, stream>>>(ei, E, N, asrc, adst, amax, denom, exbuf);

        hipMemsetAsync(bufB, 0, (size_t)N * HIDDEN * 4, stream);  // gemm already consumed bufB
        edge_scatter<<<scatBlocks, 256, 0, stream>>>(ei, E, N, exbuf, denom, bufA, bufB);
        bias_relu<<<ncBlocks, 256, 0, stream>>>(bufB, bias, N);
    }

    hipMemsetAsync(pooled, 0, (size_t)NGRAPHS * HIDDEN * 4, stream);
    hipMemsetAsync(counts, 0, (size_t)NGRAPHS * 4, stream);
    pool_sum<<<ncBlocks, 256, 0, stream>>>(bufB, batch, pooled, counts, N);
    fc_out<<<(NGRAPHS * NCLASSES + 255) / 256, 256, 0, stream>>>(pooled, counts, fc_w, fc_b, out);
    (void)out_size;
}
